// FractalTokenizer_82583631167927
// MI455X (gfx1250) — matrix operand order repaired and checked by reading, not compiled, not run on hardware
//
#include <hip/hip_runtime.h>
#include <hip/hip_bf16.h>

// ---------------------------------------------------------------------------
// FractalTokenizer forward for MI455X (gfx1250), wave32 + WMMA f16->f32.
// GEMM-shaped work (conv2/conv3 implicit GEMM, quantize, matcher, decoder)
// runs on v_wmma_f32_16x16x32_f16 with two interleaved N-tiles per K-loop
// so consecutive WMMAs are independent (avoids the 5-NOP f16 WMMA->WMMA RAW
// hazard documented in ISA 7.12.1). Weights stay f16 in L2 (192MB), conv2/3
// B-fragments stream from global; LDS is reserved for activations.
// ---------------------------------------------------------------------------

typedef __attribute__((ext_vector_type(16))) _Float16 v16h;
typedef __attribute__((ext_vector_type(8)))  _Float16 v8h;
typedef __attribute__((ext_vector_type(8)))  float    v8f;

#define NPATCH 32768   // 32 * 32 * 32 patches
#define WMMA(a,b,c) __builtin_amdgcn_wmma_f32_16x16x32_f16(false,(a),false,(b),(short)0,(c),false,false)

static __device__ __forceinline__ v16h zero_ah() { v16h z = {}; return z; }

static __device__ __forceinline__ v16h pack2(v8h lo, v8h hi) {
  return __builtin_shufflevector(lo, hi, 0,1,2,3,4,5,6,7,8,9,10,11,12,13,14,15);
}

// A fragment (16x32 f16) from row-major A[16][K]; K multiple of 32, kb mult 32.
// Lanes 0-15 hold K=kb..kb+7 & kb+16..kb+23 of row m=lane,
// lanes 16-31 hold K=kb+8..kb+15 & kb+24..kb+31 of row m=lane-16. (ISA 7.12.2)
static __device__ __forceinline__ v16h a_frag(const _Float16* A, int K, int kb, int lane) {
  int m = lane & 15, hi = lane >> 4;
  const _Float16* p = A + m * K + kb + hi * 8;
  return pack2(*(const v8h*)p, *(const v8h*)(p + 16));
}

// B fragment (32x16 f16) from row-major Wt[N][K] (torch Linear weight layout).
// Lane n = n0+(lane&15) holds column n; lanes 0-15: K=kb..kb+7 & kb+16..kb+23, 16-31: +8.
static __device__ __forceinline__ v16h b_frag(const _Float16* Wt, int K, int n0, int kb, int lane) {
  int n = n0 + (lane & 15), hi = lane >> 4;
  const _Float16* p = Wt + (long)n * K + kb + hi * 8;
  return pack2(*(const v8h*)p, *(const v8h*)(p + 16));
}

// ------------------------------- prep kernels ------------------------------

__global__ void ft_cvt_f16(const float* __restrict__ src, _Float16* __restrict__ dst, int n) {
  int i = blockIdx.x * 256 + threadIdx.x;
  if (i < n) dst[i] = (_Float16)src[i];
}

// OIHW [O][I][3][3] -> [O][kh*3+kw][I] f16  (K ordered as (tap, channel))
__global__ void ft_pack_conv(const float* __restrict__ w, _Float16* __restrict__ dst, int O, int I) {
  int i = blockIdx.x * 256 + threadIdx.x;
  int tot = O * I * 9;
  if (i < tot) {
    int o = i / (I * 9); int r = i % (I * 9);
    int ic = r / 9; int tap = r % 9;
    dst[o * (9 * I) + tap * I + ic] = (_Float16)w[i];
  }
}

// mw2 [5][256] -> [16][256] f16, rows 5..15 zero (pads N to one WMMA tile)
__global__ void ft_mw2pad(const float* __restrict__ mw2, _Float16* __restrict__ dst) {
  int i = blockIdx.x * 256 + threadIdx.x;
  if (i < 16 * 256) {
    int o = i >> 8, k = i & 255;
    dst[i] = (o < 5) ? (_Float16)mw2[o * 256 + k] : (_Float16)0.0f;
  }
}

__global__ void ft_cnorm(const float* __restrict__ cb, float* __restrict__ cn) {
  int t = threadIdx.x;                 // 256 tokens
  float s = 0.f;
  for (int j = 0; j < 256; ++j) { float v = cb[t * 256 + j]; s += v * v; }
  cn[t] = s;
}

// --------------------------- conv1 (VALU, K=27) ----------------------------
// x [32][3][256][256] fp32 -> h1 [N][64 pix][32 ch] f16, relu(conv3x3 pad1)+b1
__global__ void ft_conv1(const float* __restrict__ x, const float* __restrict__ w1,
                         const float* __restrict__ b1, _Float16* __restrict__ h1) {
  long i = (long)blockIdx.x * 256 + threadIdx.x;   // N*64*32 threads
  long n = i >> 11; int r = (int)(i & 2047);
  int pix = r >> 5, ch = r & 31;
  int py = pix >> 3, px = pix & 7;
  long b = n >> 10; int rem = (int)(n & 1023);
  int hpi = rem >> 5, wpi = rem & 31;
  const float* xb = x + ((b * 3) * 256 + (long)hpi * 8) * 256 + wpi * 8;
  float acc = b1[ch];
  for (int ic = 0; ic < 3; ++ic)
    for (int kh = 0; kh < 3; ++kh) {
      int iy = py - 1 + kh; if (iy < 0 || iy >= 8) continue;
      for (int kw = 0; kw < 3; ++kw) {
        int ix = px - 1 + kw; if (ix < 0 || ix >= 8) continue;
        acc += xb[(ic * 256 + iy) * 256 + ix] * w1[((ch * 3 + ic) * 3 + kh) * 3 + kw];
      }
    }
  if (acc < 0.f) acc = 0.f;
  h1[i] = (_Float16)acc;
}

// ----------------- conv2+conv3+pool (WMMA implicit GEMM) -------------------
// 1 wave / 4 patches, only 24KB LDS/block (high residency). conv2: M=16 px,
// K=288, N=64 (4 independent accumulators). conv3: M=16 (4pat x 4px), K=576,
// N=256 in pairs of tiles (2 independent accumulators). Weights (36KB+288KB)
// stream from L2 via b_frag; prefetched at block start.
__global__ __launch_bounds__(32) void ft_conv23(const _Float16* __restrict__ h1,
                                                const _Float16* __restrict__ w2h,
                                                const float* __restrict__ b2,
                                                const _Float16* __restrict__ w3h,
                                                const float* __restrict__ b3,
                                                float* __restrict__ z) {
  __shared__ __attribute__((aligned(128))) _Float16 h1s[4 * 64 * 32];  // 16 KB
  __shared__ __attribute__((aligned(128))) _Float16 h2s[4 * 16 * 64];  //  8 KB
  int lane = threadIdx.x;
  long pbase = (long)blockIdx.x * 4;

  // L2 prefetch of the weight streams (global_prefetch_b8)
  __builtin_prefetch((const void*)(w2h + lane * 64), 0, 1);
  __builtin_prefetch((const void*)(w3h + lane * 64), 0, 1);

  { // stage h1 (8192 halves) into LDS
    const v8h* gs = (const v8h*)(h1 + pbase * 2048);
    v8h* ld = (v8h*)h1s;
    for (int i = lane; i < 1024; i += 32) ld[i] = gs[i];
  }
  __syncthreads();

  int m = lane & 15, hi = lane >> 4;

  // ---- conv2: per patch, rows = 4x4 output pixels, cols = 64 out channels
  for (int pp = 0; pp < 4; ++pp) {
    v8f acc[4];
    v8f zf = {};
#pragma unroll
    for (int t = 0; t < 4; ++t) acc[t] = zf;
    int oy = m >> 2, ox = m & 3;
    for (int tap = 0; tap < 9; ++tap) {
      int kh = tap / 3, kw = tap % 3;
      int iy = oy * 2 - 1 + kh, ix = ox * 2 - 1 + kw;
      v16h a;
      if (iy >= 0 && iy < 8 && ix >= 0 && ix < 8) {
        const _Float16* p = h1s + ((pp << 6) + iy * 8 + ix) * 32 + hi * 8;
        a = pack2(*(const v8h*)p, *(const v8h*)(p + 16));
      } else {
        a = zero_ah();
      }
      int kb = tap * 32;
#pragma unroll
      for (int t = 0; t < 4; ++t)
        acc[t] = WMMA(a, b_frag(w2h, 288, t * 16, kb, lane), acc[t]);
    }
    // relu + bias -> h2s [pp][16 px][64 ch] f16
#pragma unroll
    for (int t = 0; t < 4; ++t) {
      int ch = t * 16 + (lane & 15);
      float bb = b2[ch];
#pragma unroll
      for (int r = 0; r < 8; ++r) {
        int mm = r + hi * 8;
        float v = acc[t][r] + bb;
        if (v < 0.f) v = 0.f;
        h2s[(pp * 16 + mm) * 64 + ch] = (_Float16)v;
      }
    }
  }
  __syncthreads();

  // ---- conv3: rows = 4 patches x 2x2 px, cols = 256, K = 9 taps x 64 ch
  int pp2 = m >> 2, pix = m & 3, oy = pix >> 1, ox = pix & 1;
  for (int t = 0; t < 16; t += 2) {    // N tiles in independent pairs
    v8f c0 = {}, c1 = {};
    for (int tap = 0; tap < 9; ++tap) {
      int kh = tap / 3, kw = tap % 3;
      int iy = oy * 2 - 1 + kh, ix = ox * 2 - 1 + kw;
      bool ok = (iy >= 0 && iy < 4 && ix >= 0 && ix < 4);
#pragma unroll
      for (int cc = 0; cc < 2; ++cc) {  // channel halves 0..31 / 32..63
        v16h a;
        if (ok) {
          const _Float16* p = h2s + (pp2 * 16 + iy * 4 + ix) * 64 + cc * 32 + hi * 8;
          a = pack2(*(const v8h*)p, *(const v8h*)(p + 16));
        } else {
          a = zero_ah();
        }
        int kb = tap * 64 + cc * 32;
        c0 = WMMA(a, b_frag(w3h, 576, t * 16, kb, lane), c0);
        c1 = WMMA(a, b_frag(w3h, 576, (t + 1) * 16, kb, lane), c1);
      }
    }
    // mean-pool 2x2 (+bias). Rows r=0..3 / 4..7 of each lane-half are one patch.
    int ch0 = t * 16 + (lane & 15), ch1 = ch0 + 16;
    float bb0 = b3[ch0], bb1 = b3[ch1];
    long p0 = (pbase + hi * 2 + 0) * 256, p1 = (pbase + hi * 2 + 1) * 256;
    z[p0 + ch0] = (c0[0] + c0[1] + c0[2] + c0[3]) * 0.25f + bb0;
    z[p1 + ch0] = (c0[4] + c0[5] + c0[6] + c0[7]) * 0.25f + bb0;
    z[p0 + ch1] = (c1[0] + c1[1] + c1[2] + c1[3]) * 0.25f + bb1;
    z[p1 + ch1] = (c1[4] + c1[5] + c1[6] + c1[7]) * 0.25f + bb1;
  }
}

// -------- quantize + matcher + decoder + scatter (WMMA everywhere) ---------
// 1 wave / 16 patches. Argmin is register-resident with a __shfl_xor
// butterfly over the 16-lane column groups of the WMMA C layout. LDS arena
// regions are phase-overlapped (57KB).
__global__ __launch_bounds__(32) void ft_tail(const float* __restrict__ z,
                                              const _Float16* __restrict__ cbh,
                                              const float* __restrict__ cnorm,
                                              const _Float16* __restrict__ mw1h,
                                              const float* __restrict__ mb1,
                                              const _Float16* __restrict__ mw2p,
                                              const float* __restrict__ mb2,
                                              const _Float16* __restrict__ dw1h,
                                              const float* __restrict__ db1,
                                              const float* __restrict__ ln_g,
                                              const float* __restrict__ ln_b,
                                              const _Float16* __restrict__ dw2h,
                                              const float* __restrict__ db2,
                                              float* __restrict__ out,
                                              float* __restrict__ tok,
                                              float* __restrict__ ifst,
                                              float* __restrict__ ifsp) {
  __shared__ __attribute__((aligned(128))) char arena[57344];
  _Float16* qh   = (_Float16*)(arena);           // [0,8192)      quantized rows f16
  float*    lg   = (float*)(arena + 8192);       // [8192,9216)   logits (dead before hbuf)
  int*      idxs = (int*)(arena + 9216);         // [9216,9280)   (dead before hbuf)
  float*    hbuf = (float*)(arena + 8192);       // [8192,40960)  decoder pre-LN f32
  _Float16* th   = (_Float16*)(arena + 40960);   // [40960,49152) matcher hidden f16
  _Float16* hh   = (_Float16*)(arena + 40960);   // [40960,57344) decoder hidden f16 (th dead)
  _Float16* zh   = (_Float16*)(arena + 49152);   // [49152,57344) z rows f16 (dead before hh)

  int lane = threadIdx.x;
  long nb = (long)blockIdx.x * 16;
  int hi = lane >> 4;

  __builtin_prefetch((const void*)(cbh + lane * 64), 0, 1);
  __builtin_prefetch((const void*)(mw1h + lane * 64), 0, 1);
  __builtin_prefetch((const void*)(dw1h + lane * 64), 0, 1);
  __builtin_prefetch((const void*)(dw2h + lane * 64), 0, 1);

  // load z rows -> f16
  for (int i = lane; i < 16 * 256; i += 32) zh[i] = (_Float16)z[nb * 256 + i];
  __syncthreads();

  // ---- distances: d2 = ||c||^2 - 2 z.c (||z||^2 constant per row), running
  // argmin in registers. Lane holds col n for rows r+hi*8.
  float best[8]; int bidx[8];
#pragma unroll
  for (int r = 0; r < 8; ++r) { best[r] = 3.4e38f; bidx[r] = 0; }
  for (int t = 0; t < 16; t += 2) {
    v8f c0 = {}, c1 = {};
#pragma unroll
    for (int k = 0; k < 8; ++k) {
      v16h a = a_frag(zh, 256, k * 32, lane);
      c0 = WMMA(a, b_frag(cbh, 256, t * 16, k * 32, lane), c0);
      c1 = WMMA(a, b_frag(cbh, 256, (t + 1) * 16, k * 32, lane), c1);
    }
    int n0 = t * 16 + (lane & 15), n1 = n0 + 16;
    float cn0 = cnorm[n0], cn1 = cnorm[n1];
#pragma unroll
    for (int r = 0; r < 8; ++r) {
      float d0 = cn0 - 2.0f * c0[r];
      float d1 = cn1 - 2.0f * c1[r];
      if (d0 < best[r]) { best[r] = d0; bidx[r] = n0; }
      if (d1 < best[r]) { best[r] = d1; bidx[r] = n1; }
    }
  }
  // butterfly min-reduce across each 16-lane column group (wave32 shuffles)
#pragma unroll
  for (int off = 8; off >= 1; off >>= 1) {
#pragma unroll
    for (int r = 0; r < 8; ++r) {
      float ov = __shfl_xor(best[r], off, 32);
      int   oi = __shfl_xor(bidx[r], off, 32);
      if (ov < best[r] || (ov == best[r] && oi < bidx[r])) { best[r] = ov; bidx[r] = oi; }
    }
  }
  if ((lane & 15) == 0) {   // lane 0 -> rows 0..7, lane 16 -> rows 8..15
#pragma unroll
    for (int r = 0; r < 8; ++r) {
      int m = r + hi * 8;
      idxs[m] = bidx[r];
      tok[nb + m] = (float)bidx[r];
    }
  }
  __syncthreads();

  // gather quantized rows (f16) from codebook
  for (int i = lane; i < 16 * 32; i += 32) {
    int m = i >> 5, cchunk = i & 31;
    *(v8h*)(qh + m * 256 + cchunk * 8) = *(const v8h*)(cbh + (long)idxs[m] * 256 + cchunk * 8);
  }
  __syncthreads();

  // ---- matcher L1: th = relu(q @ mw1^T + mb1)  [16x256], tile pairs
  for (int t = 0; t < 16; t += 2) {
    v8f c0 = {}, c1 = {};
#pragma unroll
    for (int k = 0; k < 8; ++k) {
      v16h a = a_frag(qh, 256, k * 32, lane);
      c0 = WMMA(a, b_frag(mw1h, 256, t * 16, k * 32, lane), c0);
      c1 = WMMA(a, b_frag(mw1h, 256, (t + 1) * 16, k * 32, lane), c1);
    }
    int n0 = t * 16 + (lane & 15), n1 = n0 + 16;
    float bb0 = mb1[n0], bb1 = mb1[n1];
#pragma unroll
    for (int r = 0; r < 8; ++r) {
      int mm = r + hi * 8;
      float v0 = c0[r] + bb0; if (v0 < 0.f) v0 = 0.f;
      float v1 = c1[r] + bb1; if (v1 < 0.f) v1 = 0.f;
      th[mm * 256 + n0] = (_Float16)v0;
      th[mm * 256 + n1] = (_Float16)v1;
    }
  }
  __syncthreads();

  // ---- matcher L2: logits = th @ mw2pad^T + mb2  (N padded to 16)
  {
    v8f c = {};
#pragma unroll
    for (int k = 0; k < 8; ++k)
      c = WMMA(a_frag(th, 256, k * 32, lane), b_frag(mw2p, 256, 0, k * 32, lane), c);
    int n = lane & 15;
    float bb = (n < 5) ? mb2[n] : 0.f;
#pragma unroll
    for (int r = 0; r < 8; ++r)
      lg[(r + hi * 8) * 16 + n] = c[r] + bb;
  }
  __syncthreads();

  if (lane < 16) {
    float l[5]; float mx = -1e30f; int ai = 0;
    for (int j = 0; j < 5; ++j) {
      l[j] = lg[lane * 16 + j];
      if (l[j] > mx) { mx = l[j]; ai = j; }
    }
    ifst[nb + lane] = (float)ai;
    float s = 0.f;
    for (int j = 0; j < 5; ++j) { l[j] = expf(l[j] - mx); s += l[j]; }
    float inv = 1.0f / s;
    for (int j = 0; j < 5; ++j) ifsp[(nb + lane) * 5 + j] = l[j] * inv;
  }
  __syncthreads();

  // ---- decoder L1: hbuf = q @ dw1^T + db1  [16x512], tile pairs
  for (int t = 0; t < 32; t += 2) {
    v8f c0 = {}, c1 = {};
#pragma unroll
    for (int k = 0; k < 8; ++k) {
      v16h a = a_frag(qh, 256, k * 32, lane);
      c0 = WMMA(a, b_frag(dw1h, 256, t * 16, k * 32, lane), c0);
      c1 = WMMA(a, b_frag(dw1h, 256, (t + 1) * 16, k * 32, lane), c1);
    }
    int n0 = t * 16 + (lane & 15), n1 = n0 + 16;
    float bb0 = db1[n0], bb1 = db1[n1];
#pragma unroll
    for (int r = 0; r < 8; ++r) {
      int mm = r + hi * 8;
      hbuf[mm * 512 + n0] = c0[r] + bb0;
      hbuf[mm * 512 + n1] = c1[r] + bb1;
    }
  }
  __syncthreads();

  // ---- LayerNorm(512) + ReLU -> hh f16
  if (lane < 16) {
    const float* row = hbuf + lane * 512;
    float mu = 0.f;
    for (int j = 0; j < 512; ++j) mu += row[j];
    mu *= (1.0f / 512.0f);
    float var = 0.f;
    for (int j = 0; j < 512; ++j) { float d = row[j] - mu; var += d * d; }
    var *= (1.0f / 512.0f);
    float rs = rsqrtf(var + 1e-5f);
    for (int j = 0; j < 512; ++j) {
      float v = (row[j] - mu) * rs * ln_g[j] + ln_b[j];
      if (v < 0.f) v = 0.f;
      hh[lane * 512 + j] = (_Float16)v;
    }
  }
  __syncthreads();

  // ---- decoder L2: pix = tanh(hh @ dw2^T + db2) [16x192], scatter to image
  for (int t = 0; t < 12; t += 2) {
    v8f c0 = {}, c1 = {};
#pragma unroll
    for (int k = 0; k < 16; ++k) {
      v16h a = a_frag(hh, 512, k * 32, lane);
      c0 = WMMA(a, b_frag(dw2h, 512, t * 16, k * 32, lane), c0);
      c1 = WMMA(a, b_frag(dw2h, 512, (t + 1) * 16, k * 32, lane), c1);
    }
#pragma unroll
    for (int u = 0; u < 2; ++u) {
      int j = (t + u) * 16 + (lane & 15);
      float bb = db2[j];
      int cch = j >> 6, py = (j >> 3) & 7, px = j & 7;
#pragma unroll
      for (int r = 0; r < 8; ++r) {
        long n = nb + r + hi * 8;
        long b = n >> 10; int rem = (int)(n & 1023);
        int hpi = rem >> 5, wpi = rem & 31;
        float v = (u == 0) ? c0[r] : c1[r];
        out[((b * 3 + cch) * 256 + (long)hpi * 8 + py) * 256 + wpi * 8 + px] = tanhf(v + bb);
      }
    }
  }
}

// ------------------------------ host launcher ------------------------------

extern "C" void kernel_launch(void* const* d_in, const int* in_sizes, int n_in,
                              void* d_out, int out_size, void* d_ws, size_t ws_size,
                              hipStream_t stream) {
  (void)in_sizes; (void)n_in; (void)out_size; (void)ws_size;
  const float* x   = (const float*)d_in[0];
  const float* cb  = (const float*)d_in[1];
  const float* w1  = (const float*)d_in[2];
  const float* b1  = (const float*)d_in[3];
  const float* w2  = (const float*)d_in[4];
  const float* b2  = (const float*)d_in[5];
  const float* w3  = (const float*)d_in[6];
  const float* b3  = (const float*)d_in[7];
  const float* mw1 = (const float*)d_in[8];
  const float* mb1 = (const float*)d_in[9];
  const float* mw2 = (const float*)d_in[10];
  const float* mb2 = (const float*)d_in[11];
  const float* dw1 = (const float*)d_in[12];
  const float* db1 = (const float*)d_in[13];
  const float* lng = (const float*)d_in[14];
  const float* lnb = (const float*)d_in[15];
  const float* dw2 = (const float*)d_in[16];
  const float* db2 = (const float*)d_in[17];

  // workspace carve-out (256B aligned regions)
  char* w = (char*)d_ws;
  auto carve = [&](size_t bytes) { char* p = w; w += (bytes + 255) & ~(size_t)255; return p; };
  _Float16* h1    = (_Float16*)carve((size_t)NPATCH * 64 * 32 * 2); // 128 MB
  float*    zbuf  = (float*)   carve((size_t)NPATCH * 256 * 4);     //  32 MB
  _Float16* w2h   = (_Float16*)carve(64 * 288 * 2);
  _Float16* w3h   = (_Float16*)carve(256 * 576 * 2);
  _Float16* cbh   = (_Float16*)carve(256 * 256 * 2);
  float*    cnorm = (float*)   carve(256 * 4);
  _Float16* mw1h  = (_Float16*)carve(256 * 256 * 2);
  _Float16* mw2p  = (_Float16*)carve(16 * 256 * 2);
  _Float16* dw1h  = (_Float16*)carve(512 * 256 * 2);
  _Float16* dw2h  = (_Float16*)carve(192 * 512 * 2);

  float* out  = (float*)d_out;                       // [32,3,256,256]
  float* tok  = out + 6291456;                       // token_indices [N]
  float* ifst = tok + NPATCH;                        // ifs_tokens [N]
  float* ifsp = ifst + NPATCH;                       // ifs_probs [N,5]

  // weight prep
  ft_pack_conv<<<(64 * 32 * 9 + 255) / 256, 256, 0, stream>>>(w2, w2h, 64, 32);
  ft_pack_conv<<<(256 * 64 * 9 + 255) / 256, 256, 0, stream>>>(w3, w3h, 256, 64);
  ft_cvt_f16<<<(65536 + 255) / 256, 256, 0, stream>>>(cb, cbh, 65536);
  ft_cvt_f16<<<(65536 + 255) / 256, 256, 0, stream>>>(mw1, mw1h, 65536);
  ft_cvt_f16<<<(131072 + 255) / 256, 256, 0, stream>>>(dw1, dw1h, 131072);
  ft_cvt_f16<<<(98304 + 255) / 256, 256, 0, stream>>>(dw2, dw2h, 98304);
  ft_mw2pad<<<16, 256, 0, stream>>>(mw2, mw2p);
  ft_cnorm<<<1, 256, 0, stream>>>(cb, cnorm);

  // pipeline
  ft_conv1<<<(NPATCH * 64 * 32) / 256, 256, 0, stream>>>(x, w1, b1, h1);
  ft_conv23<<<NPATCH / 4, 32, 0, stream>>>(h1, w2h, b2, w3h, b3, zbuf);
  ft_tail<<<NPATCH / 16, 32, 0, stream>>>(zbuf, cbh, cnorm, mw1h, mb1, mw2p, mb2,
                                          dw1h, db1, lng, lnb, dw2h, db2,
                                          out, tok, ifst, ifsp);
}
